// InceptionDenseGCN_7378753815017
// MI455X (gfx1250) — matrix operand order repaired and checked by reading, not compiled, not run on hardware
//
#include <hip/hip_runtime.h>

// ---------------------------------------------------------------------------
// InceptionDenseGCN for MI455X (gfx1250, wave32, WMMA + TDM)
//
// EdgeConv identity:  W.[x_i, x_j-x_i] = x_i.(W_i - W_d) + x_j.W_d
//   -> per-node GEMMs A = feats.(W_i-W_d)+b, B = feats.W_d  (7.2 GFLOP total,
//      vs 58 GFLOP in the naive per-edge formulation)
// Monotone relu:      max_j relu(A_i + B_j) = relu(A_i + max_j B_j)
//   -> edge stage is a float4 gather-max (B slice per graph = 384KB, L2-hot)
// GEMM: v_wmma_f32_16x16x32_f16; the 96 x cin f16 weight panel is DMA'd into
// LDS by the Tensor Data Mover (tensor_load_to_lds + s_wait_tensorcnt), with
// a cooperative-load fallback if the builtin is unavailable.
// ---------------------------------------------------------------------------

typedef _Float16 f16;
typedef __attribute__((ext_vector_type(16))) _Float16 v16h;
typedef __attribute__((ext_vector_type(8)))  _Float16 v8h;
typedef __attribute__((ext_vector_type(4)))  _Float16 v4h;
typedef __attribute__((ext_vector_type(8)))  float    v8f;
typedef __attribute__((ext_vector_type(4)))  float    f32x4;
typedef __attribute__((ext_vector_type(4)))  unsigned int u32x4;
typedef __attribute__((ext_vector_type(4)))  int      i32x4;
typedef __attribute__((ext_vector_type(8)))  int      i32x8;

#define NN      16384
#define CC      96
#define KNEIGH  8
#define KMAX    16
#define OUTC    192      // [A | B] concatenated output columns
#define FSTRIDE 384      // dense feature buffer width (96*4)
#define GRAPH   1024
#define NGRAPH  16
#define NEGINF  (-3.4e38f)

#ifndef __has_builtin
#define __has_builtin(x) 0
#endif
#if __has_builtin(__builtin_amdgcn_tensor_load_to_lds) && \
    __has_builtin(__builtin_amdgcn_s_wait_tensorcnt)
#define USE_TDM 1
#else
#define USE_TDM 0
#endif

union V16 { v16h v; v8h h[2]; };

__device__ __forceinline__ f32x4 vmax4(f32x4 a, f32x4 b) {
  f32x4 r;
  r.x = fmaxf(a.x, b.x); r.y = fmaxf(a.y, b.y);
  r.z = fmaxf(a.z, b.z); r.w = fmaxf(a.w, b.w);
  return r;
}

// ---- build transposed, fused weights: Wt[192][cin] (f16) ------------------
// cols 0..95  (A part): Wt[n][k] = W[k][n] - W[k+cin][n]   (= W_i - W_d)
// cols 96..191(B part): Wt[n][k] = W[k+cin][n-96]          (= W_d)
__global__ void prep_w_kernel(const float* __restrict__ w, f16* __restrict__ wt,
                              int cin) {
  int idx = blockIdx.x * blockDim.x + threadIdx.x;
  if (idx >= OUTC * cin) return;
  int n = idx / cin, k = idx % cin;
  float v;
  if (n < CC) v = w[k * CC + n] - w[(k + cin) * CC + n];
  else        v = w[(k + cin) * CC + (n - CC)];
  wt[(size_t)n * cin + k] = (f16)v;
}

// ---- init: feats[:, :96] = x (f16), res = x (covers outs[0] for both) -----
__global__ void init_kernel(const float* __restrict__ x, f16* __restrict__ feats,
                            float* __restrict__ res) {
  int idx = blockIdx.x * blockDim.x + threadIdx.x;
  if (idx >= NN * 24) return;
  int i = idx / 24, c = (idx % 24) * 4;
  f32x4 v = *(const f32x4*)&x[(size_t)i * CC + c];
  v4h hv = {(f16)v.x, (f16)v.y, (f16)v.z, (f16)v.w};
  *(v4h*)&feats[(size_t)i * FSTRIDE + c] = hv;
  *(f32x4*)&res[(size_t)i * CC + c] = v;
}

// ---- WMMA GEMM: AB[:, half*96:+96] = feats[:, :CIN] @ Wt^T (+bias on A) ---
// grid (NN/128, 2); 8 waves/block; wave owns 16 rows x 96 cols (6 tiles).
template <int CIN>
__global__ __launch_bounds__(256) void gemm_kernel(
    const f16* __restrict__ feats, const f16* __restrict__ wt,
    const float* __restrict__ bias, float* __restrict__ AB) {
  __shared__ f16 lds_w[CC * CIN];          // this block's 96 cols of W^T

  const int tid  = threadIdx.x;
  const int wave = tid >> 5;
  const int lane = tid & 31;
  const int ln   = lane & 15;              // A: M row | B/C/D: N col
  const int hh   = lane >> 4;              // lane-half selects K / M groups
  const int half = blockIdx.y;             // 0 -> A columns, 1 -> B columns
  const int row0 = blockIdx.x * 128 + wave * 16;

#if USE_TDM
  // Tensor Data Mover: DMA the contiguous 96*CIN-element f16 weight panel
  // into LDS with one descriptor (1-D tensor, data_size = 2 bytes).
  if (wave == 0) {
    const unsigned int tot = (unsigned int)(CC * CIN);   // elements (<= 27648)
    unsigned long long ga =
        (unsigned long long)(__UINTPTR_TYPE__)(wt + (size_t)half * CC * CIN);
    unsigned int lds_off = (unsigned int)(__UINTPTR_TYPE__)(&lds_w[0]);

    u32x4 g0;
    g0.x = 1u;                                           // count=1, user D#
    g0.y = lds_off;                                      // lds_addr
    g0.z = (unsigned int)(ga & 0xffffffffu);             // global_addr[31:0]
    g0.w = (unsigned int)((ga >> 32) & 0x01ffffffu) |    // global_addr[56:32]
           (2u << 30);                                   // type = 2 ("image")

    i32x8 g1;
    g1[0] = (int)(1u << 16);                             // data_size = 1 (2B)
    g1[1] = (int)((tot & 0xffffu) << 16);                // tensor_dim0 lo
    g1[2] = (int)((tot >> 16) | (1u << 16));             // dim0 hi | dim1 = 1
    g1[3] = (int)((tot & 0xffffu) << 16);                // tile_dim0 = tot
    g1[4] = 0;                                           // tile_dim1/2 unused
    g1[5] = (int)tot;                                    // dim0_stride lo
    g1[6] = 0;
    g1[7] = 0;

    i32x4 z4 = {0, 0, 0, 0};
#if __clang_major__ >= 23
    i32x8 z8 = {0, 0, 0, 0, 0, 0, 0, 0};
    __builtin_amdgcn_tensor_load_to_lds(g0, g1, z4, z4, z8, 0);
#else
    __builtin_amdgcn_tensor_load_to_lds(g0, g1, z4, z4, 0);
#endif
    __builtin_amdgcn_s_wait_tensorcnt(0);
  }
#else
  // fallback: cooperative staging of the 96 x CIN f16 weight panel
  for (int idx = tid; idx < CC * (CIN / 8); idx += 256) {
    int n = idx / (CIN / 8), q = idx % (CIN / 8);
    *(f32x4*)(&lds_w[n * CIN + q * 8]) =
        *(const f32x4*)(&wt[(size_t)(half * CC + n) * CIN + q * 8]);
  }
#endif

  const f16* arow = feats + (size_t)(row0 + ln) * FSTRIDE;
  __builtin_prefetch(arow, 0, 1);          // global_prefetch_b8 on A strip

  v8f acc[6];
  const v8f vzero = {0.f, 0.f, 0.f, 0.f, 0.f, 0.f, 0.f, 0.f};
#pragma unroll
  for (int t = 0; t < 6; ++t) acc[t] = vzero;

  __syncthreads();                         // publish W^T panel to all 8 waves

#pragma unroll
  for (int kb = 0; kb < CIN; kb += 32) {
    // A fragment (16x32 f16): halves 0-7 -> K=kb+8*hh.., 8-15 -> K=kb+16+8*hh..
    V16 a;
    a.h[0] = *(const v8h*)(arow + kb + hh * 8);
    a.h[1] = *(const v8h*)(arow + kb + 16 + hh * 8);

    // preload all 6 B fragments into independent registers, then 6 WMMAs
    V16 bf[6];
#pragma unroll
    for (int t = 0; t < 6; ++t) {
      const f16* brow = &lds_w[(t * 16 + ln) * CIN + kb + hh * 8];
      bf[t].h[0] = *(const v8h*)(brow);
      bf[t].h[1] = *(const v8h*)(brow + 16);
    }
#pragma unroll
    for (int t = 0; t < 6; ++t)
      acc[t] = __builtin_amdgcn_wmma_f32_16x16x32_f16(
          false, a.v, false, bf[t].v, (short)0, acc[t], false, false);
  }

  // epilogue: C/D layout -> col = 16t+ln, row = row0 + 8*hh + v
#pragma unroll
  for (int t = 0; t < 6; ++t) {
    int col = half * CC + t * 16 + ln;
    float bia = (half == 0) ? bias[t * 16 + ln] : 0.0f;  // bias on A cols only
#pragma unroll
    for (int v = 0; v < 8; ++v) {
      int r = row0 + hh * 8 + v;
      AB[(size_t)r * OUTC + col] = acc[t][v] + bia;
    }
  }
}

// ---- gather-max over 8 dilated neighbors, relu, fuse into res/feats -------
__global__ void edge_agg_kernel(const float* __restrict__ AB,
                                const int* __restrict__ esrc, int dil, int blk,
                                f16* __restrict__ feats, float* __restrict__ res) {
  int idx = blockIdx.x * blockDim.x + threadIdx.x;
  if (idx >= NN * 24) return;
  int i = idx / 24, c = (idx % 24) * 4;
  f32x4 a = *(const f32x4*)&AB[(size_t)i * OUTC + c];
  f32x4 bm = {NEGINF, NEGINF, NEGINF, NEGINF};
  const int* row = esrc + (size_t)i * KMAX;
#pragma unroll
  for (int j = 0; j < KNEIGH; ++j) {
    int nb = row[j * dil];                 // every dil-th stored neighbor
    bm = vmax4(bm, *(const f32x4*)&AB[(size_t)nb * OUTC + CC + c]);
  }
  f32x4 zero = {0.f, 0.f, 0.f, 0.f};
  f32x4 h = vmax4(a + bm, zero);           // = max_j relu(a + b_j)
  f32x4* rp = (f32x4*)&res[(size_t)i * CC + c];
  *rp = vmax4(*rp, h);                     // stack+max fuse (both loops)
  if (blk < 2) {                           // dense concat for next block
    v4h hv = {(f16)h.x, (f16)h.y, (f16)h.z, (f16)h.w};
    *(v4h*)&feats[(size_t)i * FSTRIDE + CC * (blk + 1) + c] = hv;
  }
}

// ---- per-graph max pool of x: two-stage deterministic reduction -----------
__global__ void pool1_kernel(const float* __restrict__ x,
                             float* __restrict__ ppart) {
  int g = blockIdx.x >> 5, s = blockIdx.x & 31, c = threadIdx.x;
  float m = NEGINF;
  int base = g * GRAPH + s * 32;
#pragma unroll 4
  for (int r = 0; r < 32; ++r)
    m = fmaxf(m, x[(size_t)(base + r) * CC + c]);
  ppart[(size_t)blockIdx.x * CC + c] = m;
}

__global__ void pool2_kernel(const float* __restrict__ ppart,
                             float* __restrict__ pool) {
  int g = blockIdx.x, c = threadIdx.x;
  float m = NEGINF;
#pragma unroll 4
  for (int s = 0; s < 32; ++s)
    m = fmaxf(m, ppart[(size_t)(g * 32 + s) * CC + c]);
  pool[g * CC + c] = m;
}

// ---- out = max(res, pool[batch]) + x --------------------------------------
__global__ void final_kernel(const float* __restrict__ res,
                             const float* __restrict__ pool,
                             const int* __restrict__ batch,
                             const float* __restrict__ x,
                             float* __restrict__ out) {
  int idx = blockIdx.x * blockDim.x + threadIdx.x;
  if (idx >= NN * 24) return;
  int i = idx / 24, c = (idx % 24) * 4;
  f32x4 r = *(const f32x4*)&res[(size_t)i * CC + c];
  f32x4 p = *(const f32x4*)&pool[(size_t)batch[i] * CC + c];
  f32x4 xv = *(const f32x4*)&x[(size_t)i * CC + c];
  *(f32x4*)&out[(size_t)i * CC + c] = vmax4(r, p) + xv;
}

extern "C" void kernel_launch(void* const* d_in, const int* in_sizes, int n_in,
                              void* d_out, int out_size, void* d_ws, size_t ws_size,
                              hipStream_t stream) {
  const float* x     = (const float*)d_in[0];
  const int*   ei    = (const int*)d_in[1];   // [2, N*KMAX]; row 0 = src ids
  const int*   batch = (const int*)d_in[2];
  const float* w[2][3];
  const float* bi[2][3];
  int p = 3;
  for (int d = 0; d < 2; ++d)
    for (int b = 0; b < 3; ++b) {
      w[d][b]  = (const float*)d_in[p++];
      bi[d][b] = (const float*)d_in[p++];
    }

  const int cins[3] = {96, 192, 288};

  // workspace carve (~32 MB)
  char* ws = (char*)d_ws;
  f16*   feats = (f16*)ws;   ws += (size_t)NN * FSTRIDE * sizeof(f16);
  float* AB    = (float*)ws; ws += (size_t)NN * OUTC * sizeof(float);
  float* res   = (float*)ws; ws += (size_t)NN * CC * sizeof(float);
  float* pool  = (float*)ws; ws += (size_t)NGRAPH * CC * sizeof(float);
  float* ppart = (float*)ws; ws += (size_t)NGRAPH * 32 * CC * sizeof(float);
  f16* wt[2][3];
  for (int d = 0; d < 2; ++d)
    for (int b = 0; b < 3; ++b) {
      wt[d][b] = (f16*)ws;
      ws += (size_t)OUTC * cins[b] * sizeof(f16);
    }

  // 1) fuse + transpose + f16-convert weights (tiny)
  for (int d = 0; d < 2; ++d)
    for (int b = 0; b < 3; ++b) {
      int tot = OUTC * cins[b];
      prep_w_kernel<<<(tot + 255) / 256, 256, 0, stream>>>(w[d][b], wt[d][b],
                                                           cins[b]);
    }

  // 2) init feats / running max
  init_kernel<<<(NN * 24 + 255) / 256, 256, 0, stream>>>(x, feats, res);

  // 3) two dilation branches x three dense blocks: WMMA GEMM + gather-max
  dim3 ggrid(NN / 128, 2);
  for (int d = 0; d < 2; ++d) {
    int dil = d + 1;   // DILATIONS = (1, 2)
    for (int b = 0; b < 3; ++b) {
      if (b == 0)
        gemm_kernel<96><<<ggrid, 256, 0, stream>>>(feats, wt[d][b], bi[d][b], AB);
      else if (b == 1)
        gemm_kernel<192><<<ggrid, 256, 0, stream>>>(feats, wt[d][b], bi[d][b], AB);
      else
        gemm_kernel<288><<<ggrid, 256, 0, stream>>>(feats, wt[d][b], bi[d][b], AB);
      edge_agg_kernel<<<(NN * 24 + 255) / 256, 256, 0, stream>>>(
          AB, ei, dil, b, feats, res);
    }
  }

  // 4) per-graph max pool (two-stage) + residual
  pool1_kernel<<<NGRAPH * 32, CC, 0, stream>>>(x, ppart);
  pool2_kernel<<<NGRAPH, CC, 0, stream>>>(ppart, pool);
  final_kernel<<<(NN * 24 + 255) / 256, 256, 0, stream>>>(res, pool, batch, x,
                                                          (float*)d_out);
}
